// GraphEmbed_67748814127344
// MI455X (gfx1250) — matrix operand order repaired and checked by reading, not compiled, not run on hardware
//
#include <hip/hip_runtime.h>
#include <hip/hip_bf16.h>

// ---------------- problem constants ----------------
#define NDIM 128
#define SDIM 64
#define BB   2048
#define IDXN 32
#define NN   (BB * IDXN)   // 65536 nodes
#define EE   262144        // directed edges before mirroring
#define LL   2

// ---------------- WMMA types ----------------
typedef __attribute__((ext_vector_type(16))) __bf16 v16bf;
typedef __attribute__((ext_vector_type(8)))  float  v8f;

union BFrag { v16bf v; unsigned int u[8]; };
union FAcc  { v8f   v; float        f[8]; };

// ---------------- helpers ----------------
__device__ __forceinline__ unsigned short f2bf(float f) {
    unsigned int u = __float_as_uint(f);
    u += 0x7FFFu + ((u >> 16) & 1u);      // round-to-nearest-even
    return (unsigned short)(u >> 16);
}
__device__ __forceinline__ unsigned int packbf(float lo, float hi) {
    return (unsigned int)f2bf(lo) | ((unsigned int)f2bf(hi) << 16);
}
// 16-bit A-matrix (16x32) K-base for VGPR v, lane-half hf (ISA 7.12.2)
__device__ __forceinline__ int kbaseA(int v, int hf) {
    return ((v < 4) ? (2 * v) : (16 + 2 * (v - 4))) + 8 * hf;
}
// 16-bit B-matrix (32x16) K-base: same K order as A (lanes 0-15 K=0..7,16..23; lanes 16-31 K=8..15,24..31)
__device__ __forceinline__ int kbaseB(int v, int hf) {
    return ((v < 4) ? (2 * v) : (16 + 2 * (v - 4))) + 8 * hf;
}
// A fragment: 16 rows x 32 K slice (chunk kc) from an LDS bf16 tile viewed as u32
__device__ __forceinline__ v16bf a_frag(const unsigned int* tile, int rowU32,
                                        int m, int kc, int hf) {
    BFrag r;
#pragma unroll
    for (int v = 0; v < 8; ++v) {
        int kb = kbaseA(v, hf) + kc * 32;          // always even
        r.u[v] = tile[m * rowU32 + (kb >> 1)];
    }
    return r.v;
}
__device__ __forceinline__ v16bf b_frag(const unsigned int* p) {
    BFrag r;
#pragma unroll
    for (int v = 0; v < 8; ++v) r.u[v] = p[v];
    return r.v;
}
__device__ __forceinline__ v8f wmma_bf16(v16bf a, v16bf b, v8f c) {
    return __builtin_amdgcn_wmma_f32_16x16x32_bf16(
        /*neg_a=*/false, a, /*neg_b=*/false, b,
        /*c_mod=*/(short)0, c, /*reuse_a=*/false, /*reuse_b=*/false);
}
__device__ __forceinline__ float sigmoidf(float x) { return 1.0f / (1.0f + expf(-x)); }

// Async DMA: global memory -> LDS, 16 bytes, per-lane addresses (ASYNCcnt).
// ldsoff = raw LDS byte offset (low 32 bits of generic shared address).
__device__ __forceinline__ void async_g2l_b128(unsigned int ldsoff, const void* gaddr) {
    asm volatile("global_load_async_to_lds_b128 %0, %1, off"
                 :: "v"(ldsoff), "v"(gaddr) : "memory");
}
__device__ __forceinline__ void wait_async0() {
    asm volatile("s_wait_asynccnt 0x0" ::: "memory");
}

// ---------------- weight fragment packing ----------------
// W: [L, Ncols, Krow] row-major (Linear weight, output-major).  We pack
// B = W^T tiles in WMMA B-fragment order:
//   out[(((l*KC + kc)*NT + nt)*32 + lane)*8 + v] = pack(W[ncol][k], W[ncol][k+1])
//   with k = kc*32 + kbaseB(v, lane>>4), ncol = nt*16 + (lane&15)
__global__ void pack_frag_kernel(const float* __restrict__ W,
                                 unsigned int* __restrict__ out,
                                 int KC, int NT, int Ncols, int Krow, int total) {
    int id = blockIdx.x * blockDim.x + threadIdx.x;
    if (id >= total) return;
    int v = id & 7;
    int lane = (id >> 3) & 31;
    int rest = id >> 8;
    int nt = rest % NT; rest /= NT;
    int kc = rest % KC;
    int l  = rest / KC;
    int hf = lane >> 4;
    int k  = kc * 32 + kbaseB(v, hf);
    int nc = nt * 16 + (lane & 15);
    const float* row = W + ((size_t)l * Ncols + nc) * Krow + k;
    out[id] = packbf(row[0], row[1]);
}

// ---------------- hidden-state init (fp32 copy + bf16 copy) ----------------
__global__ void init_kernel(const float* __restrict__ h,
                            float* __restrict__ hn_f32,
                            unsigned int* __restrict__ hn_bf, int totalPairs) {
    int id = blockIdx.x * blockDim.x + threadIdx.x;
    if (id >= totalPairs) return;
    float a = h[2 * id], b = h[2 * id + 1];
    hn_f32[2 * id]     = a;
    hn_f32[2 * id + 1] = b;
    hn_bf[id] = packbf(a, b);
}

__global__ void zero_kernel(float4* __restrict__ p, int n4) {
    int id = blockIdx.x * blockDim.x + threadIdx.x;
    if (id < n4) p[id] = make_float4(0.f, 0.f, 0.f, 0.f);
}
__global__ void copy_kernel(const float4* __restrict__ s, float4* __restrict__ d, int n4) {
    int id = blockIdx.x * blockDim.x + threadIdx.x;
    if (id < n4) d[id] = s[id];
}

// ---------------- layer kernel 1: edge messages + scatter-sum ----------------
// m_e = relu(W1 @ h[dst] + W2 @ h[src] + b);  agg[dst] += m_e
// Block = 256 threads = 8 waves.  Wave w owns output tile nt = w (16 of 128 cols).
// Weight B-fragments (8 K-chunks) stay resident in VGPRs across the tile loop.
// Feature gather uses GLOBAL_LOAD_ASYNC_TO_LDS_B128 (memory->LDS DMA, ASYNCcnt).
__global__ void msg_kernel(const unsigned int* __restrict__ hn_bf,   // [N][64] u32 (bf16 pairs)
                           const int* __restrict__ ei,               // [2][E]
                           const unsigned int* __restrict__ msgB,    // packed fragments
                           const float* __restrict__ msg_b,          // [L][128]
                           float* __restrict__ agg,                  // [N][128]
                           int layer, int numTiles) {
    __shared__ unsigned int tileA[16 * 128];   // 16 edges x 256 bf16 features
    __shared__ int dstIdx[16];
    __shared__ int srcIdx[16];

    const int tid  = threadIdx.x;
    const int lane = tid & 31;
    const int wave = tid >> 5;       // 0..7 == nt
    const int hf   = lane >> 4;
    const int m    = lane & 15;      // A row for this lane
    const int n    = lane & 15;      // C/D column for this lane

    // resident weight fragments for this wave's N-tile
    BFrag bw[8];
#pragma unroll
    for (int kc = 0; kc < 8; ++kc) {
        const unsigned int* p =
            msgB + ((((size_t)layer * 8 + kc) * 8 + wave) * 32 + lane) * 8;
#pragma unroll
        for (int v = 0; v < 8; ++v) bw[kc].u[v] = p[v];
    }
    const float bias = msg_b[layer * NDIM + wave * 16 + n];

    for (int tile = blockIdx.x; tile < numTiles; tile += gridDim.x) {
        __syncthreads();   // protect tileA/dstIdx reuse
        if (tid < 16) {
            int e = tile * 16 + tid;
            int d, s;
            if (e < EE) { s = ei[e];        d = ei[EE + e]; }
            else        { int e2 = e - EE; d = ei[e2]; s = ei[EE + e2]; }
            dstIdx[tid] = d;
            srcIdx[tid] = s;
        }
        __syncthreads();
        // gather [h_dst ; h_src] -> 16 x 128 u32 tile as 512 x 16B async DMAs
#pragma unroll
        for (int i = 0; i < 2; ++i) {
            int jj  = tid + i * 256;        // 0..511 : 16-byte chunk id
            int row = jj >> 5;              // edge row 0..15
            int c4  = jj & 31;              // 16B chunk within 128-u32 row
            int node = (c4 < 16) ? dstIdx[row] : srcIdx[row];
            int c    = (c4 < 16) ? (c4 * 4) : ((c4 - 16) * 4);
            const unsigned int* gp = hn_bf + (size_t)node * 64 + c;
            unsigned int ldsoff = (unsigned int)(size_t)&tileA[row * 128 + c4 * 4];
            async_g2l_b128(ldsoff, gp);
        }
        wait_async0();
        __syncthreads();

        FAcc acc;
#pragma unroll
        for (int r = 0; r < 8; ++r) acc.f[r] = 0.f;
#pragma unroll
        for (int kc = 0; kc < 8; ++kc) {
            v16bf a = a_frag(tileA, 128, m, kc, hf);
            acc.v = wmma_bf16(a, bw[kc].v, acc.v);
        }
        // bias + relu + scatter-add (segment_sum over random dst)
#pragma unroll
        for (int r = 0; r < 8; ++r) {
            int mrow = r + 8 * hf;
            float val = fmaxf(acc.f[r] + bias, 0.f);
            atomicAdd(&agg[(size_t)dstIdx[mrow] * NDIM + wave * 16 + n], val);
        }
    }
}

// ---------------- layer kernel 2: GRU cell ----------------
// gi = agg @ Wih^T + bih ; gh = h @ Whh^T + bhh ; fused gates.
// One block per 16-node tile; 8 waves x 3 N-groups cover N=384 for both GEMMs.
__global__ void gru_kernel(const float* __restrict__ agg,            // [N][128]
                           unsigned int* __restrict__ hn_bf,         // [N][64] u32
                           float* __restrict__ hn_f32,               // [N][128]
                           const unsigned int* __restrict__ gihB,
                           const unsigned int* __restrict__ ghhB,
                           const float* __restrict__ bih,            // [L][384]
                           const float* __restrict__ bhh,            // [L][384]
                           int layer) {
    __shared__ unsigned int aggT[16 * 64];   // 16 x 128 bf16
    __shared__ unsigned int hT[16 * 64];     // 16 x 128 bf16
    __shared__ float gi[16 * 384];
    __shared__ float gh[16 * 384];

    const int tid  = threadIdx.x;
    const int lane = tid & 31;
    const int wave = tid >> 5;
    const int hf   = lane >> 4;
    const int m    = lane & 15;
    const int n    = lane & 15;
    const int node0 = blockIdx.x * 16;

    // stage hidden-state tile via async DMA (16 rows x 64 u32 = 256 x 16B)
    {
        int row = tid >> 4;
        int c4  = tid & 15;
        const unsigned int* gp = hn_bf + (size_t)(node0 + row) * 64 + c4 * 4;
        unsigned int ldsoff = (unsigned int)(size_t)&hT[row * 64 + c4 * 4];
        async_g2l_b128(ldsoff, gp);
    }
    // stage agg tile with f32->bf16 conversion (needs VALU anyway)
#pragma unroll
    for (int i = 0; i < 4; ++i) {
        int j   = tid + i * 256;         // 0..1023
        int row = j >> 6;
        int c   = j & 63;
        const float* ap = agg + ((size_t)(node0 + row)) * NDIM + c * 2;
        aggT[row * 64 + c] = packbf(ap[0], ap[1]);
    }
    wait_async0();
    __syncthreads();

#pragma unroll
    for (int g = 0; g < 3; ++g) {
        int nt = wave + 8 * g;                      // 0..23
        // gi tile
        FAcc ci;
#pragma unroll
        for (int r = 0; r < 8; ++r) ci.f[r] = 0.f;
#pragma unroll
        for (int kc = 0; kc < 4; ++kc) {
            v16bf a = a_frag(aggT, 64, m, kc, hf);
            v16bf b = b_frag(gihB + ((((size_t)layer * 4 + kc) * 24 + nt) * 32 + lane) * 8);
            ci.v = wmma_bf16(a, b, ci.v);
        }
        float bi = bih[layer * 384 + nt * 16 + n];
#pragma unroll
        for (int r = 0; r < 8; ++r)
            gi[(r + 8 * hf) * 384 + nt * 16 + n] = ci.f[r] + bi;
        // gh tile
        FAcc ch;
#pragma unroll
        for (int r = 0; r < 8; ++r) ch.f[r] = 0.f;
#pragma unroll
        for (int kc = 0; kc < 4; ++kc) {
            v16bf a = a_frag(hT, 64, m, kc, hf);
            v16bf b = b_frag(ghhB + ((((size_t)layer * 4 + kc) * 24 + nt) * 32 + lane) * 8);
            ch.v = wmma_bf16(a, b, ch.v);
        }
        float bh = bhh[layer * 384 + nt * 16 + n];
#pragma unroll
        for (int r = 0; r < 8; ++r)
            gh[(r + 8 * hf) * 384 + nt * 16 + n] = ch.f[r] + bh;
    }
    __syncthreads();

    // fused gates, fp32
#pragma unroll
    for (int i = 0; i < 8; ++i) {
        int j  = tid + i * 256;          // 0..2047
        int mr = j >> 7;
        int d  = j & 127;
        float r  = sigmoidf(gi[mr * 384 + d]       + gh[mr * 384 + d]);
        float z  = sigmoidf(gi[mr * 384 + 128 + d] + gh[mr * 384 + 128 + d]);
        float nn = tanhf(gi[mr * 384 + 256 + d] + r * gh[mr * 384 + 256 + d]);
        size_t idx = (size_t)(node0 + mr) * NDIM + d;
        float hold = hn_f32[idx];
        float hnew = (1.0f - z) * nn + z * hold;
        hn_f32[idx] = hnew;
        ((unsigned short*)hn_bf)[idx] = f2bf(hnew);
    }
}

// ---------------- gated readout (2 GFLOP total, scalar fp32) ----------------
__global__ void readout_kernel(const float* __restrict__ hn,
                               const float* __restrict__ fmW, const float* __restrict__ fmb,
                               const float* __restrict__ gmW, const float* __restrict__ gmb,
                               const float* __restrict__ fmiW, const float* __restrict__ fmib,
                               const float* __restrict__ gmiW, const float* __restrict__ gmib,
                               float* __restrict__ outG, float* __restrict__ outGi) {
    __shared__ float hrows[IDXN * NDIM];    // 16 KB
    __shared__ float gate1[IDXN], gate2[IDXN];
    __shared__ float acc1[SDIM], acc2[SDIM];
    const int b = blockIdx.x, t = threadIdx.x;
    if (t < SDIM) { acc1[t] = 0.f; acc2[t] = 0.f; }
#pragma unroll
    for (int i = 0; i < 16; ++i) {
        int j = t + i * 256;
        hrows[j] = hn[(size_t)b * IDXN * NDIM + j];
    }
    __syncthreads();
    if (t < IDXN) {
        float s1 = gmb[0], s2 = gmib[0];
        for (int k = 0; k < NDIM; ++k) {
            float hv = hrows[t * NDIM + k];
            s1 += hv * gmW[k];
            s2 += hv * gmiW[k];
        }
        gate1[t] = sigmoidf(s1);
        gate2[t] = sigmoidf(s2);
    }
    __syncthreads();
#pragma unroll
    for (int i = 0; i < 8; ++i) {
        int j = t + i * 256;             // 0..2047 -> (node, s)
        int node = j >> 6;
        int s = j & 63;
        float d1 = fmb[s], d2 = fmib[s];
        const float* hr = &hrows[node * NDIM];
        for (int k = 0; k < NDIM; ++k) {
            float hv = hr[k];
            d1 += hv * fmW[s * NDIM + k];
            d2 += hv * fmiW[s * NDIM + k];
        }
        atomicAdd(&acc1[s], d1 * gate1[node]);
        atomicAdd(&acc2[s], d2 * gate2[node]);
    }
    __syncthreads();
    if (t < SDIM) {
        outG[(size_t)b * SDIM + t]  = acc1[t];
        outGi[(size_t)b * SDIM + t] = acc2[t];
    }
}

// ---------------- host launcher ----------------
extern "C" void kernel_launch(void* const* d_in, const int* in_sizes, int n_in,
                              void* d_out, int out_size, void* d_ws, size_t ws_size,
                              hipStream_t stream) {
    (void)in_sizes; (void)n_in; (void)out_size; (void)ws_size;
    const float* h     = (const float*)d_in[0];
    const float* msg_W = (const float*)d_in[1];
    const float* msg_b = (const float*)d_in[2];
    const float* Wih   = (const float*)d_in[3];
    const float* Whh   = (const float*)d_in[4];
    const float* bih   = (const float*)d_in[5];
    const float* bhh   = (const float*)d_in[6];
    const float* fm_W  = (const float*)d_in[7];
    const float* fm_b  = (const float*)d_in[8];
    const float* gm_W  = (const float*)d_in[9];
    const float* gm_b  = (const float*)d_in[10];
    const float* fmi_W = (const float*)d_in[11];
    const float* fmi_b = (const float*)d_in[12];
    const float* gmi_W = (const float*)d_in[13];
    const float* gmi_b = (const float*)d_in[14];
    const int*   ei    = (const int*)d_in[15];
    float* out = (float*)d_out;

    // workspace carve-out (~84.5 MB)
    char* ws = (char*)d_ws;
    float*        hn_f32 = (float*)ws;         ws += (size_t)NN * NDIM * 4;
    unsigned int* hn_bf  = (unsigned int*)ws;  ws += (size_t)NN * NDIM * 2;
    float*        agg    = (float*)ws;         ws += (size_t)NN * NDIM * 4;
    unsigned int* msgB   = (unsigned int*)ws;  ws += (size_t)2 * 8 * 8  * 256 * 4;
    unsigned int* gihB   = (unsigned int*)ws;  ws += (size_t)2 * 4 * 24 * 256 * 4;
    unsigned int* ghhB   = (unsigned int*)ws;  ws += (size_t)2 * 4 * 24 * 256 * 4;

    // 1. pack weights into WMMA B-fragment order (bf16)
    {
        int tot = 2 * 8 * 8 * 256;
        pack_frag_kernel<<<(tot + 255) / 256, 256, 0, stream>>>(msg_W, msgB, 8, 8, 128, 256, tot);
    }
    {
        int tot = 2 * 4 * 24 * 256;
        pack_frag_kernel<<<(tot + 255) / 256, 256, 0, stream>>>(Wih, gihB, 4, 24, 384, 128, tot);
        pack_frag_kernel<<<(tot + 255) / 256, 256, 0, stream>>>(Whh, ghhB, 4, 24, 384, 128, tot);
    }
    // 2. init hidden state (fp32 + bf16 copies)
    init_kernel<<<(NN * 64) / 256, 256, 0, stream>>>(h, hn_f32, hn_bf, NN * 64);

    // 3. two GNN layers
    const int numEdgeTiles = (2 * EE) / 16;   // 32768
    for (int l = 0; l < LL; ++l) {
        zero_kernel<<<(NN * NDIM / 4) / 256, 256, 0, stream>>>((float4*)agg, NN * NDIM / 4);
        msg_kernel<<<4096, 256, 0, stream>>>(hn_bf, ei, msgB, msg_b, agg, l, numEdgeTiles);
        gru_kernel<<<NN / 16, 256, 0, stream>>>(agg, hn_bf, hn_f32, gihB, ghhB, bih, bhh, l);
    }

    // 4. outputs: hn, h_G, h_G_init (concatenated)
    copy_kernel<<<(NN * NDIM / 4) / 256, 256, 0, stream>>>((const float4*)hn_f32,
                                                           (float4*)out, NN * NDIM / 4);
    readout_kernel<<<BB, 256, 0, stream>>>(hn_f32, fm_W, fm_b, gm_W, gm_b,
                                           fmi_W, fmi_b, gmi_W, gmi_b,
                                           out + (size_t)NN * NDIM,
                                           out + (size_t)NN * NDIM + (size_t)BB * SDIM);
}
